// FaceRegionGuidedCrossAttention_50818053047026
// MI455X (gfx1250) — matrix operand order repaired and checked by reading, not compiled, not run on hardware
//
#include <hip/hip_runtime.h>
#include <hip/hip_bf16.h>

// ---------------------------------------------------------------------------
// FaceRegionGuidedCrossAttention for MI455X (gfx1250, wave32, WMMA f16->f32)
//
//   E=512, H=8, HD=64, B=32, P=4096, R=64
//
// Pipeline (all on `stream`, in order):
//   0. convert Wq / Wout to f16 (weights stay L2-resident: 1 MB total)
//   1. kv_proj:   K16 (B,H,R,HD) and V16^T (B,H,HD,R) via WMMA
//   2. gemm512<A=f32,out=f16>: Q16 = visual @ Wq^T + bq
//   3. attention: scores (WMMA) -> masked softmax (shfl within 16-lane half)
//                 -> attn staged via LDS (C->A layout) -> ctx16 (WMMA)
//   4. gemm512<A=f16,out=f32>: out = ctx16 @ Wo^T + bout + visual
// ---------------------------------------------------------------------------

typedef __attribute__((ext_vector_type(16))) _Float16 v16h;
typedef __attribute__((ext_vector_type(8)))  float    v8f;

constexpr int kE  = 512;
constexpr int kH  = 8;
constexpr int kHD = 64;
constexpr int kB  = 32;
constexpr int kP  = 4096;
constexpr int kR  = 64;
constexpr int kM  = kB * kP;          // 131072 rows in the two big GEMMs

__device__ __forceinline__ v8f wmma32(v16h a, v16h b, v8f c) {
  // D = A(16x32 f16) x B(32x16 f16) + C(16x16 f32)
  return __builtin_amdgcn_wmma_f32_16x16x32_f16(
      /*neg_a=*/false, a, /*neg_b=*/false, b,
      /*c_mod=*/(short)0, c, /*reuse_a=*/false, /*reuse_b=*/false);
}

// A-fragment (16x32, f16). Caller passes p = rowBase + k0 + hid*8.
// lane L: row = L%16, elements: K = hid*8 + i (i<8) and K = 16 + hid*8 + i.
__device__ __forceinline__ v16h load_a16(const _Float16* p) {
  v16h a;
#pragma unroll
  for (int i = 0; i < 8; ++i) { a[i] = p[i]; a[i + 8] = p[i + 16]; }
  return a;
}

__device__ __forceinline__ v16h load_a32(const float* p) {
  v16h a;
#pragma unroll
  for (int i = 0; i < 8; ++i) {
    a[i]     = (_Float16)p[i];
    a[i + 8] = (_Float16)p[i + 16];
  }
  return a;
}

// B-fragment (32x16, f16), B[k][n] = W[n][k]. Caller passes
// p = W + n*ld + k0 + hid*8 ; lane L: n = L%16, K = hid*8 + i (i<8) and K = 16 + hid*8 + i.
__device__ __forceinline__ v16h load_b16(const _Float16* p) {
  v16h b;
#pragma unroll
  for (int i = 0; i < 8; ++i) { b[i] = p[i]; b[i + 8] = p[i + 16]; }
  return b;
}

__device__ __forceinline__ v16h load_b32(const float* p) {
  v16h b;
#pragma unroll
  for (int i = 0; i < 8; ++i) { b[i] = (_Float16)p[i]; b[i + 8] = (_Float16)p[i + 16]; }
  return b;
}

// ---------------------------------------------------------------------------
// Kernel 0: one-shot weight conversion f32 -> f16 (Wq = in_proj rows [0,512))
// ---------------------------------------------------------------------------
__global__ void convert_weights(const float* __restrict__ Wq,
                                const float* __restrict__ Wo,
                                _Float16* __restrict__ Wq16,
                                _Float16* __restrict__ Wo16) {
  int idx = blockIdx.x * blockDim.x + threadIdx.x;
  if (idx < kE * kE) {
    Wq16[idx] = (_Float16)Wq[idx];
    Wo16[idx] = (_Float16)Wo[idx];
  }
}

// ---------------------------------------------------------------------------
// Kernel 1: K/V projection.  One wave = one 16x16 tile of the 2048x512 GEMM.
// K stored (B,H,R,HD); V stored transposed (B,H,HD,R) so the attn B-fragments
// are two contiguous 16B loads per lane.
// ---------------------------------------------------------------------------
__global__ void __launch_bounds__(256)
kv_proj(const float* __restrict__ region, const float* __restrict__ inW,
        const float* __restrict__ inB,
        _Float16* __restrict__ K16, _Float16* __restrict__ V16t) {
  const int lane = threadIdx.x & 31, wave = threadIdx.x >> 5;
  const int ln = lane & 15, hid = lane >> 4;
  const int task  = blockIdx.x * 8 + wave;   // 8192 tasks
  const int which = task >> 12;              // 0 = K, 1 = V
  const int rem   = task & 4095;
  const int m0 = (rem >> 5) * 16;            // row in (B*R)
  const int n0 = (rem & 31) * 16;            // output feature
  const float* W    = inW + (size_t)(which + 1) * kE * kE;
  const float* bias = inB + (size_t)(which + 1) * kE;

  v8f acc;
  {
    const float bv = bias[n0 + ln];
#pragma unroll
    for (int g = 0; g < 8; ++g) acc[g] = bv;
  }
  const size_t arow = (size_t)(m0 + ln) * kE;
  const size_t brow = (size_t)(n0 + ln) * kE;
#pragma unroll 4
  for (int kc = 0; kc < 16; ++kc) {
    v16h a  = load_a32(region + arow + kc * 32 + hid * 8);
    v16h bf = load_b32(W + brow + kc * 32 + hid * 8);
    acc = wmma32(a, bf, acc);
  }
  const int n = n0 + ln, h = n >> 6, hd = n & 63;
#pragma unroll
  for (int g = 0; g < 8; ++g) {
    const int row = m0 + hid * 8 + g;        // 16-row tile never crosses batch
    const int bb = row >> 6, r = row & 63;
    if (which == 0)
      K16[(((size_t)bb * kH + h) * kR + r) * kHD + hd] = (_Float16)acc[g];
    else
      V16t[(((size_t)bb * kH + h) * kHD + hd) * kR + r] = (_Float16)acc[g];
  }
}

// ---------------------------------------------------------------------------
// Kernel 2: generic N=K=512 GEMM, C = A @ W^T + bias (+ residual).
// One wave = 32 rows x 64 cols (2 M-tiles x 4 N-tiles, 8 f32 accumulators);
// the 8 waves of a block share one 32-row A strip (L0/L2 reuse).
// ---------------------------------------------------------------------------
template <bool AF16, bool OUTF16>
__global__ void __launch_bounds__(256)
gemm512(const float* __restrict__ Af, const _Float16* __restrict__ Ah,
        const _Float16* __restrict__ W16, const float* __restrict__ bias,
        const float* __restrict__ residual,
        _Float16* __restrict__ Oh, float* __restrict__ Of) {
  const int lane = threadIdx.x & 31, wave = threadIdx.x >> 5;
  const int ln = lane & 15, hid = lane >> 4;
  const int task = blockIdx.x * 8 + wave;    // 32768 tasks
  const size_t m0 = (size_t)(task >> 3) * 32;
  const int n0 = (task & 7) * 64;

  v8f acc[2][4];
#pragma unroll
  for (int t = 0; t < 4; ++t) {
    const float bv = bias[n0 + t * 16 + ln];
#pragma unroll
    for (int g = 0; g < 8; ++g) { acc[0][t][g] = bv; acc[1][t][g] = bv; }
  }

  const size_t arow0 = (m0 + ln) * (size_t)kE;
  const size_t arow1 = (m0 + 16 + ln) * (size_t)kE;
#pragma unroll 1
  for (int kc = 0; kc < 16; ++kc) {
    const int k0 = kc * 32 + hid * 8;
    v16h a0, a1;
    if constexpr (AF16) {
      a0 = load_a16(Ah + arow0 + k0);
      a1 = load_a16(Ah + arow1 + k0);
      if (kc < 15) __builtin_prefetch(Ah + arow0 + k0 + 32, 0, 0);
    } else {
      a0 = load_a32(Af + arow0 + k0);
      a1 = load_a32(Af + arow1 + k0);
      if (kc < 15) __builtin_prefetch(Af + arow0 + k0 + 32, 0, 0);
    }
#pragma unroll
    for (int t = 0; t < 4; ++t) {
      const _Float16* bp = W16 + (size_t)(n0 + t * 16 + ln) * kE + kc * 32 + hid * 8;
      v16h bf = load_b16(bp);
      acc[0][t] = wmma32(a0, bf, acc[0][t]);
      acc[1][t] = wmma32(a1, bf, acc[1][t]);
    }
  }

#pragma unroll
  for (int i = 0; i < 2; ++i)
#pragma unroll
    for (int t = 0; t < 4; ++t)
#pragma unroll
      for (int g = 0; g < 8; ++g) {
        const size_t row = m0 + i * 16 + hid * 8 + g;
        const size_t off = row * kE + n0 + t * 16 + ln;
        if constexpr (OUTF16) Oh[off] = (_Float16)acc[i][t][g];
        else                  Of[off] = acc[i][t][g] + residual[off];
      }
}

// ---------------------------------------------------------------------------
// Kernel 3: attention.  Block = 8 waves = all 8 heads of one (b, 16-row tile)
// so the distance-mask rows are read once per block.  Softmax is done in the
// C-layout (row m lives in one 16-lane half) with shfl_xor(1,2,4,8); attn is
// re-laid out C->A through 2 KB of LDS per wave (same-wave DS ops in-order).
// ---------------------------------------------------------------------------
__global__ void __launch_bounds__(256)
attn_softmax_ctx(const _Float16* __restrict__ Q16, const float* __restrict__ dist,
                 const _Float16* __restrict__ K16, const _Float16* __restrict__ V16t,
                 _Float16* __restrict__ ctx16) {
  __shared__ _Float16 attn_s[8][16 * 64];
  const int lane = threadIdx.x & 31, wave = threadIdx.x >> 5;
  const int ln = lane & 15, hid = lane >> 4;
  const int b = blockIdx.x >> 8;
  const int p0 = (blockIdx.x & 255) * 16;
  const int h = wave;

  // Q fragments (A-matrix 16x64, K-dim = HD)
  const _Float16* Qbase = Q16 + ((size_t)b * kP + p0) * kE + h * kHD;
  const v16h aq0 = load_a16(Qbase + (size_t)ln * kE + hid * 8);
  const v16h aq1 = load_a16(Qbase + (size_t)ln * kE + 32 + hid * 8);

  const _Float16* Kh = K16 + ((size_t)(b * kH + h)) * (kR * kHD);
  const _Float16* Vh = V16t + ((size_t)(b * kH + h)) * (kHD * kR);

  // scores = Q @ K^T   (B[k=hd][n=r] = K[r][hd], row-contiguous per lane)
  v8f sacc[4];
#pragma unroll
  for (int t = 0; t < 4; ++t) {
#pragma unroll
    for (int g = 0; g < 8; ++g) sacc[t][g] = 0.0f;
    const _Float16* kp = Kh + (size_t)(t * 16 + ln) * kHD + hid * 8;
    sacc[t] = wmma32(aq0, load_b16(kp), sacc[t]);
    sacc[t] = wmma32(aq1, load_b16(kp + 32), sacc[t]);
  }

  // masked softmax over R=64; write attn (f16, A-layout source) to LDS
  _Float16* as = &attn_s[wave][0];
#pragma unroll
  for (int g = 0; g < 8; ++g) {
    const int m = hid * 8 + g;
    const float* drow = dist + ((size_t)b * kP + p0 + m) * kR + ln;
    float s[4];
#pragma unroll
    for (int t = 0; t < 4; ++t)
      s[t] = sacc[t][g] * 0.125f + (1.0f - drow[t * 16]) * -1.0e9f;
    float rmax = fmaxf(fmaxf(s[0], s[1]), fmaxf(s[2], s[3]));
#pragma unroll
    for (int off = 8; off >= 1; off >>= 1)
      rmax = fmaxf(rmax, __shfl_xor(rmax, off, 32));
    float p[4], rsum = 0.0f;
#pragma unroll
    for (int t = 0; t < 4; ++t) { p[t] = __expf(s[t] - rmax); rsum += p[t]; }
#pragma unroll
    for (int off = 8; off >= 1; off >>= 1)
      rsum += __shfl_xor(rsum, off, 32);
    const float inv = 1.0f / rsum;
#pragma unroll
    for (int t = 0; t < 4; ++t)
      as[m * 64 + t * 16 + ln] = (_Float16)(p[t] * inv);
  }

  // ctx = attn @ V   (B[k=r][n=hd] = V^T[hd][r], row-contiguous per lane)
  const v16h aa0 = load_a16(as + (size_t)ln * 64 + hid * 8);
  const v16h aa1 = load_a16(as + (size_t)ln * 64 + 32 + hid * 8);
  _Float16* cbase = ctx16 + ((size_t)b * kP + p0) * kE + h * kHD;
#pragma unroll
  for (int t = 0; t < 4; ++t) {
    v8f cacc;
#pragma unroll
    for (int g = 0; g < 8; ++g) cacc[g] = 0.0f;
    const _Float16* vp = Vh + (size_t)(t * 16 + ln) * kR + hid * 8;
    cacc = wmma32(aa0, load_b16(vp), cacc);
    cacc = wmma32(aa1, load_b16(vp + 32), cacc);
#pragma unroll
    for (int g = 0; g < 8; ++g)
      cbase[(size_t)(hid * 8 + g) * kE + t * 16 + ln] = (_Float16)cacc[g];
  }
}

// ---------------------------------------------------------------------------
extern "C" void kernel_launch(void* const* d_in, const int* in_sizes, int n_in,
                              void* d_out, int out_size, void* d_ws, size_t ws_size,
                              hipStream_t stream) {
  const float* visual = (const float*)d_in[0];   // (B,P,E)
  const float* region = (const float*)d_in[1];   // (B,R,E)
  const float* dist   = (const float*)d_in[2];   // (B,P,R)
  const float* inW    = (const float*)d_in[3];   // (3E,E)
  const float* inB    = (const float*)d_in[4];   // (3E)
  const float* outW   = (const float*)d_in[5];   // (E,E)
  const float* outB   = (const float*)d_in[6];   // (E)
  float* out = (float*)d_out;

  char* ws = (char*)d_ws;
  size_t off = 0;
  _Float16* Q16   = (_Float16*)(ws + off); off += (size_t)kM * kE * 2;       // 128 MB
  _Float16* ctx16 = (_Float16*)(ws + off); off += (size_t)kM * kE * 2;       // 128 MB
  _Float16* Wq16  = (_Float16*)(ws + off); off += (size_t)kE * kE * 2;
  _Float16* Wo16  = (_Float16*)(ws + off); off += (size_t)kE * kE * 2;
  _Float16* K16   = (_Float16*)(ws + off); off += (size_t)kB * kH * kR * kHD * 2;
  _Float16* V16t  = (_Float16*)(ws + off); off += (size_t)kB * kH * kHD * kR * 2;
  (void)ws_size; (void)in_sizes; (void)n_in; (void)out_size;

  const dim3 blk(256);
  convert_weights<<<(kE * kE + 255) / 256, blk, 0, stream>>>(inW, outW, Wq16, Wo16);
  kv_proj<<<1024, blk, 0, stream>>>(region, inW, inB, K16, V16t);
  // Q = visual @ Wq^T + bq   (A = f32 visual, out = f16)
  gemm512<false, true><<<kM / 32 * 8 / 8, blk, 0, stream>>>(
      visual, nullptr, Wq16, inB, nullptr, Q16, nullptr);
  attn_softmax_ctx<<<kB * (kP / 16), blk, 0, stream>>>(Q16, dist, K16, V16t, ctx16);
  // out = ctx @ Wo^T + bout + visual   (A = f16 ctx, out = f32)
  gemm512<true, false><<<kM / 32 * 8 / 8, blk, 0, stream>>>(
      nullptr, ctx16, Wo16, outB, visual, nullptr, out);
}
